// EncoderDecoder_67800353735232
// MI455X (gfx1250) — matrix operand order repaired and checked by reading, not compiled, not run on hardware
//
#include <hip/hip_runtime.h>
#include <hip/hip_bf16.h>
#include <math.h>

// ---------------------------------------------------------------------------
// Seq2seq LSTM encoder/decoder for MI455X (gfx1250).
// - all weights packed once to bf16 in WMMA B-fragment order (L2-resident)
// - every GEMM runs on v_wmma_f32_16x16x32_bf16 with fp32 accumulation
// - weight stream double-buffered through LDS with GLOBAL_LOAD_ASYNC_TO_LDS
//   (ASYNCcnt pipeline), one n-tile per wave, both m-tiles per wave
// ---------------------------------------------------------------------------

typedef __bf16 bf16_t;
typedef __attribute__((ext_vector_type(16))) __bf16 v16bf;
typedef __attribute__((ext_vector_type(8)))  __bf16 v8bf;
typedef __attribute__((ext_vector_type(8)))  float  v8f;

#define HDIM 1024
#define VOCAB 32000
#define BATCH 32
#define SEQ 256
#define TDEC 64

// Issue two async 16B-per-lane copies: 1KB fragment block -> LDS buffer.
// Identity copy: lane l moves bytes [l*16, l*16+16) and [512+l*16, ...).
#define ASYNC_B128x2(ldsaddr, gaddr)                                        \
    asm volatile("global_load_async_to_lds_b128 %0, %1, off\n\t"            \
                 "global_load_async_to_lds_b128 %0, %1, off offset:512"     \
                 :: "v"(ldsaddr), "v"(gaddr) : "memory")

// ---------------------------------------------------------------------------
// Pack fp32 weights -> bf16 in WMMA B-fragment order.
//   lane l: n = ntile*16 + (l&15), k = kstep*32 + (l>=16 ? 16 : 0) + e, e=0..15
// Wa covers k < Khalf (row stride Khalf), Wb covers k >= Khalf.
// ---------------------------------------------------------------------------
__global__ void pack_w_kernel(const float* __restrict__ Wa,
                              const float* __restrict__ Wb,
                              bf16_t* __restrict__ dst,
                              int N, int Ktot, int Khalf)
{
    size_t total = (size_t)N * (size_t)Ktot;
    int ksteps = Ktot >> 5;
    for (size_t o = (size_t)blockIdx.x * blockDim.x + threadIdx.x;
         o < total;
         o += (size_t)gridDim.x * blockDim.x) {
        int e = (int)(o & 15);
        size_t t = o >> 4;
        int l = (int)(t & 31);
        t >>= 5;
        int kstep = (int)(t % ksteps);
        int ntile = (int)(t / ksteps);
        int n = ntile * 16 + (l & 15);
        int k = kstep * 32 + ((l >= 16) ? 16 : 0) + e;
        float v;
        if (k < Khalf) v = Wa[(size_t)n * Khalf + k];
        else           v = Wb[(size_t)n * (Ktot - Khalf) + (k - Khalf)];
        dst[o] = (bf16_t)v;
    }
}

// A-fragment loader (16-bit A-matrix 16x32 layout): two 16B loads at +0,+8.
__device__ __forceinline__ v16bf load_afrag(const bf16_t* ap)
{
    v8bf lo = *(const v8bf*)(ap);
    v8bf hi = *(const v8bf*)(ap + 8);
    v16bf a;
#pragma unroll
    for (int i = 0; i < 8; ++i) { a[i] = lo[i]; a[8 + i] = hi[i]; }
    return a;
}

// ---------------------------------------------------------------------------
// WMMA GEMM: out[32,N] = A[32,Ktot] * W^T (+ bias0 + bias1)
// A split in two halves: A0 covers k<Khalf, A1 the rest (may be null).
// W is bf16 in fragment order from pack_w_kernel.
// Block = 128 threads = 4 waves; each wave: 1 n-tile, both m-tiles.
// B-fragments double-buffered via async global->LDS copies (ASYNCcnt).
// ---------------------------------------------------------------------------
__global__ __launch_bounds__(128)
void wmma_gemm_kernel(const bf16_t* __restrict__ A0,
                      const bf16_t* __restrict__ A1,
                      const bf16_t* __restrict__ W,
                      const float*  __restrict__ bias0,
                      const float*  __restrict__ bias1,
                      float* __restrict__ out,
                      int N, int Ktot, int Khalf)
{
    __shared__ bf16_t bbuf[4][2][512];            // per-wave double buffer, 8KB

    const int lane  = threadIdx.x & 31;
    const int wave  = threadIdx.x >> 5;           // 0..3
    const int ntile = blockIdx.x * 4 + wave;
    const int row16 = lane & 15;
    const bool hi   = lane >= 16;
    const int aoff  = hi ? 16 : 0;

    const int ksteps = Ktot >> 5;
    // byte base of this n-tile's fragment stream (1KB per k-step)
    const char* wbytes = (const char*)W + (size_t)ntile * ksteps * 1024;

    const unsigned ldsblk = (unsigned)(size_t)&bbuf[wave][0][0];
    const unsigned ldsl   = (unsigned)(lane * 16);   // per-lane async offset

    // prologue: prefetch k-step 0 into buffer 0
    ASYNC_B128x2(ldsblk + ldsl, wbytes + lane * 16);

    const int   Krem = Ktot - Khalf;
    const bf16_t* Asec = A1 ? A1 : A0;
    const bf16_t* ap0 = A0 + (size_t)row16 * Khalf + aoff;
    const bf16_t* ap1 = A0 + (size_t)(row16 + 16) * Khalf + aoff;

    v8f acc0 = {}, acc1 = {};
    int ks = 0;
    for (int seg = 0; seg < 2; ++seg) {
        const int end = (seg == 0) ? (Khalf >> 5) : ksteps;
        for (; ks < end; ++ks) {
            const int nks = ks + 1;
            if (nks < ksteps) {
                ASYNC_B128x2(ldsblk + ((unsigned)(nks & 1) << 10) + ldsl,
                             wbytes + (size_t)nks * 1024 + lane * 16);
                asm volatile("s_wait_asynccnt 0x2" ::: "memory");
            } else {
                asm volatile("s_wait_asynccnt 0x0" ::: "memory");
            }
            v16bf bfrag = *(const v16bf*)((const char*)&bbuf[wave][ks & 1][0]
                                          + lane * 32);
            v16bf a0 = load_afrag(ap0);
            v16bf a1 = load_afrag(ap1);
            acc0 = __builtin_amdgcn_wmma_f32_16x16x32_bf16(
                       false, a0, false, bfrag, (short)0, acc0, false, false);
            acc1 = __builtin_amdgcn_wmma_f32_16x16x32_bf16(
                       false, a1, false, bfrag, (short)0, acc1, false, false);
            ap0 += 32; ap1 += 32;
        }
        // switch A source to second half (h part)
        const size_t stride = Krem > 0 ? (size_t)Krem : 1;
        ap0 = Asec + (size_t)row16 * stride + aoff;
        ap1 = Asec + (size_t)(row16 + 16) * stride + aoff;
    }

    // Epilogue: C/D layout -> VGPR r: M = r + (hi?8:0), N-col = row16
    const int j = ntile * 16 + row16;
    float bsum = 0.f;
    if (bias0) bsum += bias0[j];
    if (bias1) bsum += bias1[j];
    const int mA = hi ? 8 : 0;
#pragma unroll
    for (int r = 0; r < 8; ++r) {
        out[(size_t)(mA + r) * N + j]      = acc0[r] + bsum;
        out[(size_t)(16 + mA + r) * N + j] = acc1[r] + bsum;
    }
}

// ---------------------------------------------------------------------------
// LSTM elementwise: gates[32,4096] (i,f,g,o) -> c (fp32, in place), h (bf16).
// ---------------------------------------------------------------------------
__global__ void lstm_ew_kernel(const float* __restrict__ gates,
                               float* __restrict__ c,
                               bf16_t* __restrict__ h,
                               bf16_t* __restrict__ h2)
{
    int idx = blockIdx.x * blockDim.x + threadIdx.x;   // 0 .. 32*1024-1
    int b = idx >> 10, m = idx & 1023;
    const float* g = gates + (size_t)b * 4096;
    float gi = g[m], gf = g[m + 1024], gg = g[m + 2048], go = g[m + 3072];
    float si = 1.f / (1.f + __expf(-gi));
    float sf = 1.f / (1.f + __expf(-gf));
    float so = 1.f / (1.f + __expf(-go));
    float cn = sf * c[idx] + si * tanhf(gg);
    c[idx] = cn;
    bf16_t hn = (bf16_t)(so * tanhf(cn));
    h[idx] = hn;
    if (h2) h2[idx] = hn;
}

// ---------------------------------------------------------------------------
// Embedding gather -> bf16 stage buffer [32,1024]; token = toks[b*tstride].
// ---------------------------------------------------------------------------
__global__ void embed_kernel(const float* __restrict__ emb,
                             const int* __restrict__ toks,
                             int tstride,
                             bf16_t* __restrict__ dst)
{
    int idx = blockIdx.x * blockDim.x + threadIdx.x;   // 0 .. 32*1024-1
    int b = idx >> 10, k = idx & 1023;
    int tok = toks[(size_t)b * tstride];
    dst[idx] = (bf16_t)emb[(size_t)tok * HDIM + k];
}

// ---------------------------------------------------------------------------
// Per-row argmax over logits[32,V] + bias; first-index tie-break.
// ---------------------------------------------------------------------------
__global__ void argmax_kernel(const float* __restrict__ logits,
                              const float* __restrict__ bias,
                              int* __restrict__ tok,
                              int* __restrict__ outcol)
{
    __shared__ float sv[256];
    __shared__ int   si[256];
    const int b = blockIdx.x;
    const float* row = logits + (size_t)b * VOCAB;
    float best = -INFINITY; int bi = 0;
    for (int n = threadIdx.x; n < VOCAB; n += blockDim.x) {
        float v = row[n] + bias[n];
        if (v > best) { best = v; bi = n; }
    }
    sv[threadIdx.x] = best; si[threadIdx.x] = bi;
    __syncthreads();
    for (int s = 128; s > 0; s >>= 1) {
        if (threadIdx.x < s) {
            float v2 = sv[threadIdx.x + s]; int i2 = si[threadIdx.x + s];
            if (v2 > sv[threadIdx.x] ||
                (v2 == sv[threadIdx.x] && i2 < si[threadIdx.x])) {
                sv[threadIdx.x] = v2; si[threadIdx.x] = i2;
            }
        }
        __syncthreads();
    }
    if (threadIdx.x == 0) {
        tok[b] = si[0];
        outcol[(size_t)b * TDEC] = si[0];
    }
}

// ---------------------------------------------------------------------------
__global__ void init_state_kernel(bf16_t* h0, bf16_t* h1,
                                  float* c0, float* c1, int* tok)
{
    int idx = blockIdx.x * blockDim.x + threadIdx.x;
    if (idx < BATCH * HDIM) {
        h0[idx] = (bf16_t)0.f; h1[idx] = (bf16_t)0.f;
        c0[idx] = 0.f; c1[idx] = 0.f;
    }
    if (idx < BATCH) tok[idx] = 1;   // BOS
}

// ---------------------------------------------------------------------------
extern "C" void kernel_launch(void* const* d_in, const int* in_sizes, int n_in,
                              void* d_out, int out_size, void* d_ws, size_t ws_size,
                              hipStream_t stream)
{
    const int*   x       = (const int*)  d_in[0];
    const float* enc_emb = (const float*)d_in[1];
    const float* dec_emb = (const float*)d_in[2];
    const float* eWih0 = (const float*)d_in[3];
    const float* eWhh0 = (const float*)d_in[4];
    const float* ebih0 = (const float*)d_in[5];
    const float* ebhh0 = (const float*)d_in[6];
    const float* eWih1 = (const float*)d_in[7];
    const float* eWhh1 = (const float*)d_in[8];
    const float* ebih1 = (const float*)d_in[9];
    const float* ebhh1 = (const float*)d_in[10];
    const float* dWih0 = (const float*)d_in[11];
    const float* dWhh0 = (const float*)d_in[12];
    const float* dbih0 = (const float*)d_in[13];
    const float* dbhh0 = (const float*)d_in[14];
    const float* dWih1 = (const float*)d_in[15];
    const float* dWhh1 = (const float*)d_in[16];
    const float* dbih1 = (const float*)d_in[17];
    const float* dbhh1 = (const float*)d_in[18];
    const float* predW = (const float*)d_in[19];
    const float* predb = (const float*)d_in[20];

    // ---- workspace carve-up ----
    char* p = (char*)d_ws;
    auto carve = [&](size_t bytes) -> char* {
        char* r = p;
        p += (bytes + 255) & ~(size_t)255;
        return r;
    };
    const size_t cellW = (size_t)4096 * 2048 * sizeof(bf16_t);   // 16.78 MB
    bf16_t* WcE0   = (bf16_t*)carve(cellW);
    bf16_t* WcE1   = (bf16_t*)carve(cellW);
    bf16_t* WcD0   = (bf16_t*)carve(cellW);
    bf16_t* WcD1   = (bf16_t*)carve(cellW);
    bf16_t* Wpred  = (bf16_t*)carve((size_t)VOCAB * HDIM * sizeof(bf16_t)); // 65.5 MB
    bf16_t* hs0    = (bf16_t*)carve((size_t)SEQ * BATCH * HDIM * sizeof(bf16_t));
    bf16_t* xstage = (bf16_t*)carve((size_t)BATCH * HDIM * sizeof(bf16_t));
    bf16_t* h0bf   = (bf16_t*)carve((size_t)BATCH * HDIM * sizeof(bf16_t));
    bf16_t* h1bf   = (bf16_t*)carve((size_t)BATCH * HDIM * sizeof(bf16_t));
    float*  c0     = (float*) carve((size_t)BATCH * HDIM * sizeof(float));
    float*  c1     = (float*) carve((size_t)BATCH * HDIM * sizeof(float));
    float*  gates  = (float*) carve((size_t)BATCH * 4096 * sizeof(float));
    float*  logits = (float*) carve((size_t)BATCH * VOCAB * sizeof(float));
    int*    tok    = (int*)   carve((size_t)BATCH * sizeof(int));

    // ---- one-time per call: pack weights to bf16 fragment order ----
    pack_w_kernel<<<2048, 256, 0, stream>>>(eWih0, eWhh0, WcE0, 4096, 2048, 1024);
    pack_w_kernel<<<2048, 256, 0, stream>>>(eWih1, eWhh1, WcE1, 4096, 2048, 1024);
    pack_w_kernel<<<2048, 256, 0, stream>>>(dWih0, dWhh0, WcD0, 4096, 2048, 1024);
    pack_w_kernel<<<2048, 256, 0, stream>>>(dWih1, dWhh1, WcD1, 4096, 2048, 1024);
    pack_w_kernel<<<4096, 256, 0, stream>>>(predW, nullptr, Wpred, VOCAB, 1024, 1024);

    init_state_kernel<<<(BATCH * HDIM + 255) / 256, 256, 0, stream>>>(h0bf, h1bf, c0, c1, tok);

    const int ewBlocks  = (BATCH * HDIM) / 256;   // 128
    const int gemmCell  = 4096 / 64;              // 64 blocks x 4 waves
    const int gemmVocab = VOCAB / 64;             // 500 blocks x 4 waves

    // ---- encoder layer 0 (sequential scan) ----
    for (int t = 0; t < SEQ; ++t) {
        embed_kernel<<<ewBlocks, 256, 0, stream>>>(enc_emb, x + t, SEQ, xstage);
        wmma_gemm_kernel<<<gemmCell, 128, 0, stream>>>(
            xstage, h0bf, WcE0, ebih0, ebhh0, gates, 4096, 2048, 1024);
        lstm_ew_kernel<<<ewBlocks, 256, 0, stream>>>(
            gates, c0, h0bf, hs0 + (size_t)t * BATCH * HDIM);
    }
    // ---- encoder layer 1 ----
    for (int t = 0; t < SEQ; ++t) {
        wmma_gemm_kernel<<<gemmCell, 128, 0, stream>>>(
            hs0 + (size_t)t * BATCH * HDIM, h1bf, WcE1, ebih1, ebhh1,
            gates, 4096, 2048, 1024);
        lstm_ew_kernel<<<ewBlocks, 256, 0, stream>>>(gates, c1, h1bf, nullptr);
    }
    // ---- greedy autoregressive decoder ----
    for (int t = 0; t < TDEC; ++t) {
        embed_kernel<<<ewBlocks, 256, 0, stream>>>(dec_emb, tok, 1, xstage);
        wmma_gemm_kernel<<<gemmCell, 128, 0, stream>>>(
            xstage, h0bf, WcD0, dbih0, dbhh0, gates, 4096, 2048, 1024);
        lstm_ew_kernel<<<ewBlocks, 256, 0, stream>>>(gates, c0, h0bf, nullptr);
        wmma_gemm_kernel<<<gemmCell, 128, 0, stream>>>(
            h0bf, h1bf, WcD1, dbih1, dbhh1, gates, 4096, 2048, 1024);
        lstm_ew_kernel<<<ewBlocks, 256, 0, stream>>>(gates, c1, h1bf, nullptr);
        wmma_gemm_kernel<<<gemmVocab, 128, 0, stream>>>(
            h1bf, nullptr, Wpred, nullptr, nullptr, logits, VOCAB, 1024, 1024);
        argmax_kernel<<<BATCH, 256, 0, stream>>>(logits, predb, tok, (int*)d_out + t);
    }
}
